// ptBEVnet_51427938403103
// MI455X (gfx1250) — compile-verified, matrix operand order checked by reading, not run on hardware
//
#include <hip/hip_runtime.h>

// ---------------- problem constants ----------------
constexpr int   Bsz   = 2;
constexpr int   Npt   = 100000;
constexpr int   FEA   = 9;
constexpr int   GXg   = 480;
constexpr int   GYg   = 360;
constexpr int   GHg   = 32;
constexpr int   DOUTc = 512;
constexpr int   MAXPT = 256;
constexpr float EPSc  = 1e-5f;
constexpr float SLOPEc= 0.01f;

constexpr int NSEG  = Bsz * GXg * GYg;   // 345600 voxels
constexpr int NPTS  = Bsz * Npt;         // 200000 points
constexpr int NTILE = NPTS / 16;         // 12500 16-point tiles (exact)
constexpr int VTILE = NSEG / 16;         // 21600 16-voxel tiles (exact)
constexpr int LSTR  = 264;               // LDS row stride (ushorts), 16B-multiple, != 256 to stagger banks

// bf16 B-fragment table: layer -> (fragment offset, Kt, Nt)
constexpr int FOFF1 = 0;    // 9->64   : Kt=1,  Nt=4   (4 frags)
constexpr int FOFF2 = 4;    // 64->128 : Kt=2,  Nt=8   (16)
constexpr int FOFF3 = 20;   // 128->256: Kt=4,  Nt=16  (64)
constexpr int FOFF4 = 84;   // 256->512: Kt=8,  Nt=32  (256)
constexpr int FOFF5 = 340;  // 512->32 : Kt=16, Nt=2   (32)
constexpr int NFRAG = 372;

// k3 LDS: 2 waves x (ping+pong) x 32 rows x LSTR ushorts
constexpr unsigned K3_SMEM = 2u * 2u * 32u * LSTR * 2u;   // 67,584 bytes (dynamic)

// ---------------- WMMA types ----------------
typedef __attribute__((ext_vector_type(16))) __bf16 v16bf;
typedef __attribute__((ext_vector_type(8)))  float  v8f;

union Frag {
    v16bf v;
    uint4 q[2];
    unsigned short s[16];
};

__device__ __forceinline__ v8f wmma_bf16(const Frag& a, const Frag& b, v8f c) {
    // 8 args: (neg_a, A, neg_b, B, c_mod, C, reuse_a, reuse_b)
    return __builtin_amdgcn_wmma_f32_16x16x32_bf16(false, a.v, false, b.v, (short)0, c, false, false);
}

__device__ __forceinline__ unsigned short f2bfb(float f) { // float -> bf16 bits (RNE)
    unsigned u = __float_as_uint(f);
    u += 0x7FFFu + ((u >> 16) & 1u);
    return (unsigned short)(u >> 16);
}
// order-preserving float<->int encoding so atomicMax(int) == float max
__device__ __forceinline__ int   encf(float f) { int i = __float_as_int(f); return i >= 0 ? i : (i ^ 0x7FFFFFFF); }
__device__ __forceinline__ float decf(int e)   { return __int_as_float(e >= 0 ? e : (e ^ 0x7FFFFFFF)); }

__device__ __forceinline__ float lrelu(float x) { return x > 0.0f ? x : SLOPEc * x; }

// load one pre-swizzled B fragment: 32 bytes per lane, fully coalesced
__device__ __forceinline__ void loadW(Frag& b, const unsigned short* __restrict__ wfrag, int f, int lane) {
    const uint4* p = (const uint4*)(wfrag + ((size_t)f * 32 + lane) * 16);
    b.q[0] = p[0];
    b.q[1] = p[1];
}
// load A fragment from an LDS activation row (two ds_load_b128)
__device__ __forceinline__ void loadA(Frag& a, const unsigned short* row, int kt, int kbase) {
    a.q[0] = *(const uint4*)(row + kt * 32 + kbase);
    a.q[1] = *(const uint4*)(row + kt * 32 + 16 + kbase);
}

// ---------------- k0: init max-feature encodings + counts ----------------
__global__ void k0_init(int* __restrict__ maxenc, int* __restrict__ counts) {
    const long long i = (long long)blockIdx.x * blockDim.x + threadIdx.x;
    const long long tot4 = (long long)NSEG * 128;         // NSEG*512 / 4
    if (i < tot4) {
        const int NEGI = (int)0x80000000;                 // <= encf(-inf): neutral for max
        int4 v; v.x = NEGI; v.y = NEGI; v.z = NEGI; v.w = NEGI;
        *(int4*)(maxenc + i * 4) = v;
    }
    if (i < NSEG) counts[i] = 0;
}

// ---------------- k1: voxel id + per-voxel rank (cap at MAX_PT) ----------------
__global__ void k1_rank(const int* __restrict__ x_ind, const int* __restrict__ y_ind,
                        int* __restrict__ counts, int* __restrict__ keptvox) {
    const int i = blockIdx.x * blockDim.x + threadIdx.x;
    if (i >= NPTS) return;
    const int b   = i / Npt;
    const int vox = (b * GXg + x_ind[i]) * GYg + y_ind[i];
    const int rank = atomicAdd(&counts[vox], 1);
    keptvox[i] = (rank < MAXPT) ? vox : -1;
}

// ---------------- k2: fold BN into weights, swizzle to WMMA B-fragment layout ----------------
__global__ void k2_wprep(const float* __restrict__ W1, const float* __restrict__ W2,
                         const float* __restrict__ W3, const float* __restrict__ W4,
                         const float* __restrict__ W5,
                         const float* __restrict__ bn1g, const float* __restrict__ bn1v,
                         const float* __restrict__ bn2g, const float* __restrict__ bn2v,
                         const float* __restrict__ bn3g, const float* __restrict__ bn3v,
                         unsigned short* __restrict__ wfrag) {
    const int f = blockIdx.x;          // fragment id, 32 threads per block
    const int lane = threadIdx.x;
    const float* W; const float* sg = nullptr; const float* sv = nullptr;
    int Kreal, Kt, Dout, rel;
    if      (f < FOFF2) { W = W1; Kreal = 9;   Kt = 1;  Dout = 64;  rel = f;         sg = bn1g; sv = bn1v; }
    else if (f < FOFF3) { W = W2; Kreal = 64;  Kt = 2;  Dout = 128; rel = f - FOFF2; sg = bn2g; sv = bn2v; }
    else if (f < FOFF4) { W = W3; Kreal = 128; Kt = 4;  Dout = 256; rel = f - FOFF3; sg = bn3g; sv = bn3v; }
    else if (f < FOFF5) { W = W4; Kreal = 256; Kt = 8;  Dout = 512; rel = f - FOFF4; }
    else                { W = W5; Kreal = 512; Kt = 16; Dout = 32;  rel = f - FOFF5; }
    const int nt = rel / Kt, kt = rel % Kt;
    const int lo = lane & 15, hi = lane >> 4;
    const int n  = nt * 16 + lo;
    const float s = sg ? sg[n] * rsqrtf(sv[n] + EPSc) : 1.0f;
    unsigned short* dst = wfrag + ((size_t)f * 32 + lane) * 16;
    #pragma unroll
    for (int e = 0; e < 16; ++e) {
        const int k = kt * 32 + (e < 8 ? hi * 8 + e : 16 + hi * 8 + (e - 8));
        const float val = (k < Kreal) ? W[(size_t)k * Dout + n] * s : 0.0f;
        dst[e] = f2bfb(val);
    }
}

// folded biases: fb = (b - bn_m)*s + bn_b ; fb1@0(64) fb2@64(128) fb3@192(256)
__global__ void k2_bias(const float* __restrict__ b1, const float* __restrict__ b2, const float* __restrict__ b3,
                        const float* g1, const float* bb1, const float* m1, const float* v1,
                        const float* g2, const float* bb2, const float* m2, const float* v2,
                        const float* g3, const float* bb3, const float* m3, const float* v3,
                        float* __restrict__ fbias) {
    const int i = blockIdx.x * blockDim.x + threadIdx.x;
    if (i < 64) {
        const float s = g1[i] * rsqrtf(v1[i] + EPSc);
        fbias[i] = (b1[i] - m1[i]) * s + bb1[i];
    } else if (i < 192) {
        const int n = i - 64; const float s = g2[n] * rsqrtf(v2[n] + EPSc);
        fbias[i] = (b2[n] - m2[n]) * s + bb2[n];
    } else if (i < 448) {
        const int n = i - 192; const float s = g3[n] * rsqrtf(v3[n] + EPSc);
        fbias[i] = (b3[n] - m3[n]) * s + bb3[n];
    }
}

// ---------------- k3: fused point MLP, M=32 per wave (2 A-tiles share each B fragment) ----------------
__global__ __launch_bounds__(64) void k3_pointnet(
        const float* __restrict__ pt_fea,
        const float* __restrict__ bn0g, const float* __restrict__ bn0b,
        const float* __restrict__ bn0m, const float* __restrict__ bn0v,
        const float* __restrict__ b4,
        const int*   __restrict__ keptvox,
        const unsigned short* __restrict__ wfrag,
        const float* __restrict__ fbias,
        int* __restrict__ maxenc) {
    extern __shared__ __align__(16) unsigned char k3_smem_raw[];
    unsigned short* smem = (unsigned short*)k3_smem_raw;

    const int lane = threadIdx.x & 31;
    const int w    = threadIdx.x >> 5;
    const int lo   = lane & 15, hi = lane >> 4;
    const int pair = blockIdx.x * 2 + w;           // < 6250 exactly
    const int t0   = pair * 2;                     // first of two 16-point tiles

    unsigned short* actA = smem + (size_t)w * (2u * 32u * LSTR);   // 32 rows (2 M-tiles)
    unsigned short* actB = actA + 32u * LSTR;

    // ---- per-lane BN0 scale/offset for the k's this lane covers ----
    float sc[8], of[8];
    #pragma unroll
    for (int e = 0; e < 8; ++e) { sc[e] = 0.0f; of[e] = 0.0f; }
    if (hi == 0) {
        #pragma unroll
        for (int k = 0; k < 8; ++k) {
            const float s = bn0g[k] * rsqrtf(bn0v[k] + EPSc);
            sc[k] = s; of[k] = bn0b[k] - bn0m[k] * s;
        }
    } else {
        const float s = bn0g[8] * rsqrtf(bn0v[8] + EPSc);
        sc[0] = s; of[0] = bn0b[8] - bn0m[8] * s;    // k = 8
    }

    // ---- layer-1 A fragments: BN0(pt_fea), K padded 9 -> 32 ----
    Frag a1[2];
    #pragma unroll
    for (int ta = 0; ta < 2; ++ta) {
        #pragma unroll
        for (int e = 0; e < 16; ++e) a1[ta].s[e] = 0;
        const int p = (t0 + ta) * 16 + lo;
        const float* fr = pt_fea + (size_t)p * FEA;
        if (hi == 0) {
            #pragma unroll
            for (int k = 0; k < 8; ++k) a1[ta].s[k] = f2bfb(fr[k] * sc[k] + of[k]);
        } else {
            a1[ta].s[0] = f2bfb(fr[8] * sc[0] + of[0]);
        }
    }

    // ---- layer 1: [32x32] x [32x64] -> actA ----
    #pragma unroll
    for (int nt = 0; nt < 4; ++nt) {
        Frag b; loadW(b, wfrag, FOFF1 + nt, lane);
        const int n = nt * 16 + lo;
        const float bias = fbias[n];
        #pragma unroll
        for (int ta = 0; ta < 2; ++ta) {
            v8f acc = {};
            acc = wmma_bf16(a1[ta], b, acc);
            #pragma unroll
            for (int r = 0; r < 8; ++r)
                actA[(ta * 16 + r + 8 * hi) * LSTR + n] = f2bfb(lrelu(acc[r] + bias));
        }
    }
    __syncthreads();

    // ---- layer 2: [32x64] x [64x128] -> actB ----
    {
        Frag a[2][2];
        #pragma unroll
        for (int ta = 0; ta < 2; ++ta)
            #pragma unroll
            for (int kt = 0; kt < 2; ++kt)
                loadA(a[ta][kt], actA + (ta * 16 + lo) * LSTR, kt, 8 * hi);
        for (int nt = 0; nt < 8; ++nt) {
            v8f acc[2] = { v8f{}, v8f{} };
            #pragma unroll
            for (int kt = 0; kt < 2; ++kt) {
                Frag b; loadW(b, wfrag, FOFF2 + nt * 2 + kt, lane);
                #pragma unroll
                for (int ta = 0; ta < 2; ++ta) acc[ta] = wmma_bf16(a[ta][kt], b, acc[ta]);
            }
            const int n = nt * 16 + lo;
            const float bias = fbias[64 + n];
            #pragma unroll
            for (int ta = 0; ta < 2; ++ta)
                #pragma unroll
                for (int r = 0; r < 8; ++r)
                    actB[(ta * 16 + r + 8 * hi) * LSTR + n] = f2bfb(lrelu(acc[ta][r] + bias));
        }
    }
    __syncthreads();

    // ---- layer 3: [32x128] x [128x256] -> actA ----
    {
        Frag a[2][4];
        #pragma unroll
        for (int ta = 0; ta < 2; ++ta)
            #pragma unroll
            for (int kt = 0; kt < 4; ++kt)
                loadA(a[ta][kt], actB + (ta * 16 + lo) * LSTR, kt, 8 * hi);
        for (int nt = 0; nt < 16; ++nt) {
            v8f acc[2] = { v8f{}, v8f{} };
            #pragma unroll
            for (int kt = 0; kt < 4; ++kt) {
                Frag b; loadW(b, wfrag, FOFF3 + nt * 4 + kt, lane);
                #pragma unroll
                for (int ta = 0; ta < 2; ++ta) acc[ta] = wmma_bf16(a[ta][kt], b, acc[ta]);
            }
            const int n = nt * 16 + lo;
            const float bias = fbias[192 + n];
            #pragma unroll
            for (int ta = 0; ta < 2; ++ta)
                #pragma unroll
                for (int r = 0; r < 8; ++r)
                    actA[(ta * 16 + r + 8 * hi) * LSTR + n] = f2bfb(lrelu(acc[ta][r] + bias));
        }
    }
    __syncthreads();

    // ---- layer 4: [32x256] x [256x512] -> atomic max scatter to voxels ----
    {
        Frag a[2][8];
        #pragma unroll
        for (int ta = 0; ta < 2; ++ta)
            #pragma unroll
            for (int kt = 0; kt < 8; ++kt)
                loadA(a[ta][kt], actA + (ta * 16 + lo) * LSTR, kt, 8 * hi);
        int kv[2][8];
        #pragma unroll
        for (int ta = 0; ta < 2; ++ta)
            #pragma unroll
            for (int r = 0; r < 8; ++r)
                kv[ta][r] = keptvox[(t0 + ta) * 16 + r + 8 * hi];

        for (int nt = 0; nt < 32; ++nt) {
            // gfx1250 prefetch of the next n-tile's fragment block (global_prefetch_b8)
            if (nt < 31)
                __builtin_prefetch((const void*)(wfrag + ((size_t)(FOFF4 + (nt + 1) * 8) * 32 + lane) * 16), 0, 1);
            v8f acc[2] = { v8f{}, v8f{} };
            #pragma unroll
            for (int kt = 0; kt < 8; ++kt) {
                Frag b; loadW(b, wfrag, FOFF4 + nt * 8 + kt, lane);
                #pragma unroll
                for (int ta = 0; ta < 2; ++ta) acc[ta] = wmma_bf16(a[ta][kt], b, acc[ta]);
            }
            const int n = nt * 16 + lo;
            const float bias = b4[n];
            #pragma unroll
            for (int ta = 0; ta < 2; ++ta)
                #pragma unroll
                for (int r = 0; r < 8; ++r) {
                    if (kv[ta][r] >= 0)
                        atomicMax(&maxenc[(size_t)kv[ta][r] * DOUTc + n], encf(acc[ta][r] + bias));
                }
        }
    }
}

// ---------------- k4: voxel MLP [16x512] x [512x32] (WMMA) -> dense BEV output ----------------
__global__ __launch_bounds__(64) void k4_voxel(
        const int* __restrict__ maxenc, const int* __restrict__ counts,
        const unsigned short* __restrict__ wfrag, const float* __restrict__ b5,
        float* __restrict__ out) {
    const int lane = threadIdx.x & 31;
    const int w    = threadIdx.x >> 5;
    const int lo   = lane & 15, hi = lane >> 4;
    const int tile = blockIdx.x * 2 + w;           // < 21600 exactly
    const int vrow = tile * 16 + lo;               // voxel for this lane's A row
    const bool rowOcc = counts[vrow] > 0;
    const int* src = maxenc + (size_t)vrow * DOUTc;

    v8f acc[2] = { v8f{}, v8f{} };
    for (int kt = 0; kt < 16; ++kt) {
        Frag a;
        const int k0 = kt * 32 + 8 * hi;
        #pragma unroll
        for (int e = 0; e < 8; ++e) {
            a.s[e]     = rowOcc ? f2bfb(decf(src[k0 + e]))      : (unsigned short)0;
            a.s[e + 8] = rowOcc ? f2bfb(decf(src[k0 + 16 + e])) : (unsigned short)0;
        }
        Frag b0, b1;
        loadW(b0, wfrag, FOFF5 + kt,      lane);   // nt=0
        loadW(b1, wfrag, FOFF5 + 16 + kt, lane);   // nt=1
        acc[0] = wmma_bf16(a, b0, acc[0]);
        acc[1] = wmma_bf16(a, b1, acc[1]);
    }

    int cnt[8];
    #pragma unroll
    for (int r = 0; r < 8; ++r) cnt[r] = counts[tile * 16 + r + 8 * hi];

    #pragma unroll
    for (int nt = 0; nt < 2; ++nt) {
        const int c = nt * 16 + lo;                // output channel
        const float bias = b5[c];
        #pragma unroll
        for (int r = 0; r < 8; ++r) {
            const int vv = tile * 16 + r + 8 * hi;
            const float val = (cnt[r] > 0) ? lrelu(acc[nt][r] + bias) : 0.0f;
            const int b  = vv / (GXg * GYg);
            const int rm = vv - b * (GXg * GYg);
            const int xg = rm / GYg;
            const int yg = rm - xg * GYg;
            out[(((size_t)b * GHg + c) * GXg + xg) * GYg + yg] = val;
        }
    }
}

// ---------------- host launcher ----------------
extern "C" void kernel_launch(void* const* d_in, const int* in_sizes, int n_in,
                              void* d_out, int out_size, void* d_ws, size_t ws_size,
                              hipStream_t stream) {
    (void)in_sizes; (void)n_in; (void)out_size; (void)ws_size;
    const float* pt_fea = (const float*)d_in[0];
    const int*   x_ind  = (const int*)  d_in[1];
    const int*   y_ind  = (const int*)  d_in[2];
    const float* bn0g = (const float*)d_in[3],  *bn0b = (const float*)d_in[4];
    const float* bn0m = (const float*)d_in[5],  *bn0v = (const float*)d_in[6];
    const float* bn1g = (const float*)d_in[7],  *bn1b = (const float*)d_in[8];
    const float* bn1m = (const float*)d_in[9],  *bn1v = (const float*)d_in[10];
    const float* bn2g = (const float*)d_in[11], *bn2b = (const float*)d_in[12];
    const float* bn2m = (const float*)d_in[13], *bn2v = (const float*)d_in[14];
    const float* bn3g = (const float*)d_in[15], *bn3b = (const float*)d_in[16];
    const float* bn3m = (const float*)d_in[17], *bn3v = (const float*)d_in[18];
    const float* W1 = (const float*)d_in[19], *b1 = (const float*)d_in[20];
    const float* W2 = (const float*)d_in[21], *b2 = (const float*)d_in[22];
    const float* W3 = (const float*)d_in[23], *b3 = (const float*)d_in[24];
    const float* W4 = (const float*)d_in[25], *b4 = (const float*)d_in[26];
    const float* W5 = (const float*)d_in[27], *b5 = (const float*)d_in[28];

    // workspace carve-up (all 16B-aligned offsets)
    char* ws = (char*)d_ws;
    int*            counts  = (int*)ws;                                           // NSEG ints
    int*            keptvox = (int*)(ws + (size_t)NSEG * 4);                      // NPTS ints
    unsigned short* wfrag   = (unsigned short*)(ws + (size_t)NSEG * 4 + (size_t)NPTS * 4);
    float*          fbias   = (float*)((char*)wfrag + (size_t)NFRAG * 32 * 16 * 2);
    int*            maxenc  = (int*)((char*)fbias + 512 * 4);                     // NSEG*512 ints

    const long long initThreads = (long long)NSEG * 128;      // int4 per thread
    k0_init   <<<(unsigned)((initThreads + 255) / 256), 256, 0, stream>>>(maxenc, counts);
    k1_rank   <<<(NPTS + 255) / 256, 256, 0, stream>>>(x_ind, y_ind, counts, keptvox);
    k2_wprep  <<<NFRAG, 32, 0, stream>>>(W1, W2, W3, W4, W5, bn1g, bn1v, bn2g, bn2v, bn3g, bn3v, wfrag);
    k2_bias   <<<2, 256, 0, stream>>>(b1, b2, b3,
                                      bn1g, bn1b, bn1m, bn1v,
                                      bn2g, bn2b, bn2m, bn2v,
                                      bn3g, bn3b, bn3m, bn3v, fbias);
    k3_pointnet<<<NTILE / 4, 64, K3_SMEM, stream>>>(pt_fea, bn0g, bn0b, bn0m, bn0v,
                                                    b4, keptvox, wfrag, fbias, maxenc);
    k4_voxel  <<<VTILE / 2, 64, 0, stream>>>(maxenc, counts, wfrag, b5, (float*)d_out);
}
